// SelfAttention_13718125543473
// MI455X (gfx1250) — matrix operand order repaired and checked by reading, not compiled, not run on hardware
//
#include <hip/hip_runtime.h>
#include <hip/hip_bf16.h>
#include <math.h>

#define BATCH 64
#define SEQ   2048
#define DIM   1024
#define UNITS 1024

typedef __attribute__((ext_vector_type(16))) __bf16 v16bf;
typedef __attribute__((ext_vector_type(8)))  __bf16 v8bf;
typedef __attribute__((ext_vector_type(8)))  float  v8f;

#define ROWS 128         // rows of hidden per block in the scores kernel
#define LDA  1032        // bf16 elements per LDS row (1024 + 8 pad -> 16B pad)

#if __has_builtin(__builtin_amdgcn_tanhf)
#define TANH(x) __builtin_amdgcn_tanhf(x)    // v_tanh_f32 on gfx1250
#else
#define TANH(x) tanhf(x)
#endif

// ---------------------------------------------------------------------------
// Kernel 1: wsu[b,u] = sum_d s_prev[b,d]*W[d,u] + W_bias[u] + U_bias[u]
// ---------------------------------------------------------------------------
__global__ __launch_bounds__(256) void k_wsu(const float* __restrict__ s_prev,
                                             const float* __restrict__ W,
                                             const float* __restrict__ Wb,
                                             const float* __restrict__ Ub,
                                             float* __restrict__ wsu) {
  const int g = blockIdx.x * 256 + threadIdx.x;     // 64*1024 outputs
  const int b = g >> 10;
  const int u = g & 1023;
  const float* sp = s_prev + (size_t)b * DIM;
  float acc = Wb[u] + Ub[u];
#pragma unroll 4
  for (int d = 0; d < DIM; ++d)
    acc = fmaf(sp[d], W[(size_t)d * UNITS + u], acc);
  wsu[g] = acc;
}

// ---------------------------------------------------------------------------
// Kernel 2: Ut[u,d] = (bf16) U[d,u]  (tiled transpose for coalescing)
// ---------------------------------------------------------------------------
__global__ __launch_bounds__(256) void k_ut(const float* __restrict__ U,
                                            __bf16* __restrict__ Ut) {
  __shared__ float tile[32][33];
  const int tx = threadIdx.x & 31;
  const int ty = threadIdx.x >> 5;      // 0..7
  const int d0 = blockIdx.x * 32;
  const int u0 = blockIdx.y * 32;
#pragma unroll
  for (int i = 0; i < 4; ++i) {
    int d = d0 + ty + i * 8;
    tile[ty + i * 8][tx] = U[(size_t)d * UNITS + (u0 + tx)];
  }
  __syncthreads();
#pragma unroll
  for (int i = 0; i < 4; ++i) {
    int u = u0 + ty + i * 8;
    Ut[(size_t)u * DIM + (d0 + tx)] = (__bf16)tile[tx][ty + i * 8];
  }
}

// ---------------------------------------------------------------------------
// Kernel 3: fused  scores[m] = V_bias + sum_n tanh(wsu[b,n] + (h@U)[m,n]) * V[n]
// bf16 WMMA GEMM, register double-buffered, tanh+V-dot epilogue kept in regs.
// Each of the 8 waves owns a 16-row sub-tile and 2 N sub-tiles per chunk.
// ---------------------------------------------------------------------------
union Frag { v16bf v; v8bf h[2]; };

#define LOAD_A(dst, kk)                                                  \
  dst.h[0] = *(const v8bf*)(arow + (kk) + half8);                        \
  dst.h[1] = *(const v8bf*)(arow + (kk) + half8 + 16);

#define LOAD_B(dst, brow, kk)                                            \
  dst.h[0] = *(const v8bf*)((brow) + (kk) + half8);                      \
  dst.h[1] = *(const v8bf*)((brow) + (kk) + half8 + 16);

__global__ __launch_bounds__(256) void k_scores(const float* __restrict__ hs,
                                                const float* __restrict__ wsu,
                                                const __bf16* __restrict__ Ut,
                                                const float* __restrict__ Vk,
                                                const float* __restrict__ Vb,
                                                float* __restrict__ scores) {
  extern __shared__ char smem[];
  __bf16* Ah   = (__bf16*)smem;                             // ROWS x LDA bf16
  float*  wsuL = (float*)(smem + (size_t)ROWS * LDA * 2);   // 1024 f32
  float*  vL   = wsuL + UNITS;                              // 1024 f32

  const int tid = threadIdx.x;
  const int m0  = blockIdx.x * ROWS;                  // global row base
  const int b   = m0 / SEQ;                           // ROWS | SEQ -> uniform
  const float vbias = Vb[0];

  for (int i = tid; i < UNITS; i += 256) {
    wsuL[i] = wsu[(size_t)b * UNITS + i];
    vL[i]   = Vk[i];
  }

  // Stage hidden tile (ROWS x DIM fp32) into LDS as bf16, 16B-padded rows.
  const float* hbase = hs + (size_t)m0 * DIM;
  for (int idx = tid; idx < ROWS * DIM / 4; idx += 256) {
    int r  = idx >> 8;                                // 256 float4 per row
    int c4 = idx & 255;
    float4 f = ((const float4*)(hbase + (size_t)r * DIM))[c4];
    __bf16* dst = Ah + r * LDA + c4 * 4;
    dst[0] = (__bf16)f.x; dst[1] = (__bf16)f.y;
    dst[2] = (__bf16)f.z; dst[3] = (__bf16)f.w;
  }
  __syncthreads();

  const int wave   = tid >> 5;                        // 0..7
  const int lane   = tid & 31;
  const int half   = lane >> 4;
  const int l16    = lane & 15;
  const int half8  = half * 8;
  const int half16 = half * 16;
  const int mrow   = wave * 16;                       // wave-exclusive rows

  const __bf16* arow = Ah + (size_t)(mrow + l16) * LDA;   // A: lane = M row

  float psum[8];
#pragma unroll
  for (int r = 0; r < 8; ++r) psum[r] = 0.f;

  for (int nc = 0; nc < UNITS; nc += 32) {
    const int n0 = nc + l16;                          // lane's column, tile 0
    const int n1 = nc + 16 + l16;                     // lane's column, tile 1
    const __bf16* brow0 = Ut + (size_t)n0 * DIM;      // B^T rows: contiguous K
    const __bf16* brow1 = Ut + (size_t)n1 * DIM;

    v8f acc0 = {}, acc1 = {};
    Frag a0, a1, b00, b01, b10, b11;
    LOAD_A(a0, 0) LOAD_B(b00, brow0, 0) LOAD_B(b10, brow1, 0)

#pragma unroll 2
    for (int kk = 0; kk < DIM; kk += 64) {
      const int k1 = kk + 32;
      LOAD_A(a1, k1) LOAD_B(b01, brow0, k1) LOAD_B(b11, brow1, k1)
      acc0 = __builtin_amdgcn_wmma_f32_16x16x32_bf16(
          false, a0.v, false, b00.v, (short)0, acc0, false, false);
      acc1 = __builtin_amdgcn_wmma_f32_16x16x32_bf16(
          false, a0.v, false, b10.v, (short)0, acc1, false, false);
      const int k2 = (kk + 64) & (DIM - 1);
      LOAD_A(a0, k2) LOAD_B(b00, brow0, k2) LOAD_B(b10, brow1, k2)
      acc0 = __builtin_amdgcn_wmma_f32_16x16x32_bf16(
          false, a1.v, false, b01.v, (short)0, acc0, false, false);
      acc1 = __builtin_amdgcn_wmma_f32_16x16x32_bf16(
          false, a1.v, false, b11.v, (short)0, acc1, false, false);
    }

    // Epilogue: accumulate tanh(u_h + wsu) * V into per-lane partials.
    const float wn0 = wsuL[n0], vn0 = vL[n0];
    const float wn1 = wsuL[n1], vn1 = vL[n1];
#pragma unroll
    for (int r = 0; r < 8; ++r) {
      psum[r] += TANH(acc0[r] + wn0) * vn0 + TANH(acc1[r] + wn1) * vn1;
    }
  }

  // One reduction at the end: sum over N lanes (16-lane groups), write scores.
#pragma unroll
  for (int r = 0; r < 8; ++r) {
    float p = psum[r];
    p += __shfl_xor(p, 1, 16);
    p += __shfl_xor(p, 2, 16);
    p += __shfl_xor(p, 4, 16);
    p += __shfl_xor(p, 8, 16);
    if (l16 == 0) scores[m0 + mrow + r + half * 8] = p + vbias;
  }
}

// ---------------------------------------------------------------------------
// Kernel 4: softmax over S per batch
// ---------------------------------------------------------------------------
__global__ __launch_bounds__(256) void k_softmax(const float* __restrict__ scores,
                                                 float* __restrict__ weight) {
  __shared__ float red[256];
  const int b = blockIdx.x;
  const int tid = threadIdx.x;
  const float* s = scores + (size_t)b * SEQ;

  float mx = -INFINITY;
  for (int i = tid; i < SEQ; i += 256) mx = fmaxf(mx, s[i]);
  red[tid] = mx; __syncthreads();
  for (int o = 128; o > 0; o >>= 1) {
    if (tid < o) red[tid] = fmaxf(red[tid], red[tid + o]);
    __syncthreads();
  }
  mx = red[0]; __syncthreads();

  float sum = 0.f;
  for (int i = tid; i < SEQ; i += 256) sum += __expf(s[i] - mx);
  red[tid] = sum; __syncthreads();
  for (int o = 128; o > 0; o >>= 1) {
    if (tid < o) red[tid] += red[tid + o];
    __syncthreads();
  }
  const float inv = 1.0f / red[0];
  for (int i = tid; i < SEQ; i += 256)
    weight[(size_t)b * SEQ + i] = __expf(s[i] - mx) * inv;
}

// ---------------------------------------------------------------------------
// Kernel 5: context[b,d] = sum_s h[b,s,d] * weight[b,s]   (bandwidth bound)
// ---------------------------------------------------------------------------
__global__ __launch_bounds__(256) void k_context(const float* __restrict__ hs,
                                                 const float* __restrict__ weight,
                                                 float* __restrict__ ctx) {
  __shared__ float wsh[256];
  const int b = blockIdx.y;
  const int d = blockIdx.x * 256 + threadIdx.x;
  const float* hb = hs + (size_t)b * SEQ * DIM;
  const float* wb = weight + (size_t)b * SEQ;
  float acc = 0.f;
  for (int s0 = 0; s0 < SEQ; s0 += 256) {
    wsh[threadIdx.x] = wb[s0 + threadIdx.x];
    __syncthreads();
#pragma unroll 4
    for (int s = 0; s < 256; ++s)
      acc = fmaf(hb[(size_t)(s0 + s) * DIM + d], wsh[s], acc);
    __syncthreads();
  }
  ctx[(size_t)b * DIM + d] = acc;
}

// ---------------------------------------------------------------------------
extern "C" void kernel_launch(void* const* d_in, const int* in_sizes, int n_in,
                              void* d_out, int out_size, void* d_ws, size_t ws_size,
                              hipStream_t stream) {
  const float* s_prev = (const float*)d_in[0];
  const float* hidden = (const float*)d_in[1];
  const float* Wk     = (const float*)d_in[2];
  const float* Wb     = (const float*)d_in[3];
  const float* Uk     = (const float*)d_in[4];
  const float* Ub     = (const float*)d_in[5];
  const float* Vk     = (const float*)d_in[6];
  const float* Vb     = (const float*)d_in[7];

  float* out      = (float*)d_out;
  float* ctx_out  = out;                        // [B, D]
  float* wgt_out  = out + (size_t)BATCH * DIM;  // [B, S]

  // workspace layout
  char* ws = (char*)d_ws;
  float*  wsu    = (float*)ws;                                   // B*UNITS f32
  float*  scores = (float*)(ws + (size_t)BATCH * UNITS * 4);     // B*S f32
  __bf16* Ut     = (__bf16*)(ws + (size_t)BATCH * UNITS * 4
                                + (size_t)BATCH * SEQ * 4);      // UNITS*D bf16

  k_wsu<<<(BATCH * UNITS) / 256, 256, 0, stream>>>(s_prev, Wk, Wb, Ub, wsu);
  k_ut<<<dim3(DIM / 32, UNITS / 32), 256, 0, stream>>>(Uk, Ut);

  const size_t smem = (size_t)ROWS * LDA * 2 + 2 * UNITS * 4;
  k_scores<<<(BATCH * SEQ) / ROWS, 256, smem, stream>>>(hidden, wsu, Ut, Vk, Vb,
                                                        scores);
  k_softmax<<<BATCH, 256, 0, stream>>>(scores, wgt_out);
  k_context<<<dim3(DIM / 256, BATCH), 256, 0, stream>>>(hidden, wgt_out, ctx_out);
}
